// Bottleneck_3624952397873
// MI455X (gfx1250) — matrix operand order repaired and checked by reading, not compiled, not run on hardware
//
#include <hip/hip_runtime.h>

// ---------------------------------------------------------------------------
// ResNet bottleneck on MI455X (gfx1250): wave32, v_wmma_f32_16x16x32_bf16,
// double-buffered LDS pipeline, async global->LDS A-tile staging via
// global_load_async_to_lds_b128 (ASYNCcnt path).
// ---------------------------------------------------------------------------

typedef __attribute__((ext_vector_type(16))) __bf16 v16bf;
typedef __attribute__((ext_vector_type(8)))  float  v8f;
typedef int v4i __attribute__((vector_size(16)));
typedef unsigned short u16;
typedef unsigned int   u32;

#define NB     64
#define CIN    512
#define CB     256
#define COUT   1024
#define HW_IN  28
#define HW_OUT 14
#define NPIX   (NB * HW_OUT * HW_OUT)   // 12544
#define LDP    40                       // LDS row pitch (halfwords): 80B, 16B-aligned

// ---- gfx1250 async global->LDS path (builtin confirmed to exist) ----------
#ifndef HAVE_ASYNC_LDS
#  if defined(__has_builtin)
#    if __has_builtin(__builtin_amdgcn_global_load_async_to_lds_b128)
#      define HAVE_ASYNC_LDS 1
#    endif
#  endif
#endif
#ifndef HAVE_ASYNC_LDS
#  define HAVE_ASYNC_LDS 0
#endif

typedef __attribute__((address_space(1))) v4i* gptr_v4i;
typedef __attribute__((address_space(3))) v4i* lptr_v4i;

__device__ __forceinline__ void async_wait() {
#if HAVE_ASYNC_LDS
#  if __has_builtin(__builtin_amdgcn_s_wait_asynccnt)
    __builtin_amdgcn_s_wait_asynccnt(0);
#  else
    asm volatile("s_wait_asynccnt 0x0" ::: "memory");
#  endif
#endif
}

__device__ __forceinline__ u16 f2bf(float f) {
    union { float f; u32 u; } c; c.f = f;
    u32 u = c.u;
    return (u16)((u + 0x7FFFu + ((u >> 16) & 1u)) >> 16);   // RNE
}

// ------------------------- prep kernels ------------------------------------

__global__ void scale_w_kernel(const float* __restrict__ w, const float* __restrict__ s,
                               u16* __restrict__ dst, int rows, int cols) {
    int idx = blockIdx.x * 256 + threadIdx.x;
    if (idx >= rows * cols) return;
    int row = idx / cols;
    dst[idx] = f2bf(w[idx] * s[row]);
}

__global__ void gather_x_kernel(const float* __restrict__ x, u16* __restrict__ xs) {
    int idx = blockIdx.x * 256 + threadIdx.x;
    if (idx >= CIN * NPIX) return;
    int ci = idx / NPIX, p = idx % NPIX;
    int n = p / 196, hw = p % 196, h = hw / 14, w = hw % 14;
    xs[idx] = f2bf(x[(size_t)n * (CIN * HW_IN * HW_IN) + ci * (HW_IN * HW_IN)
                    + (2 * h) * HW_IN + 2 * w]);
}

// ------------------------- GEMM building blocks ----------------------------

// A tile 128x32: issue either async global->LDS copies, or global loads to regs.
__device__ __forceinline__ void issue_A(u16* AsBuf, const u16* __restrict__ gA,
                                        int K, int m0, int k0, int tid, uint4* aR) {
    int row = tid >> 1;                 // 0..127
    int ch  = (tid & 1) * 16;           // 0 or 16
    const u16* src = gA + (size_t)(m0 + row) * K + k0 + ch;
#if HAVE_ASYNC_LDS
    u16* dst = AsBuf + row * LDP + ch;
    __builtin_amdgcn_global_load_async_to_lds_b128(
        (gptr_v4i)(src), (lptr_v4i)(dst), 0, 0);
    __builtin_amdgcn_global_load_async_to_lds_b128(
        (gptr_v4i)(src + 8), (lptr_v4i)(dst + 8), 0, 0);
    (void)aR;
#else
    aR[0] = ((const uint4*)src)[0];
    aR[1] = ((const uint4*)src)[1];
#endif
}

__device__ __forceinline__ void commit_A(u16* AsBuf, int tid, const uint4* aR) {
#if !HAVE_ASYNC_LDS
    int row = tid >> 1;
    int ch  = (tid & 1) * 16;
    uint4* dst = reinterpret_cast<uint4*>(AsBuf + row * LDP + ch);
    dst[0] = aR[0];
    dst[1] = aR[1];
#else
    (void)AsBuf; (void)tid; (void)aR;
#endif
}

// Dense B tile 32x64 (row-major [K][NPIX]): load 8 bf16/thread into a reg.
__device__ __forceinline__ uint4 issue_B_dense(const u16* __restrict__ gB,
                                               int k0, int n0, int tid) {
    int krow = tid >> 3;
    int nseg = (tid & 7) * 8;
    return *reinterpret_cast<const uint4*>(gB + (size_t)(k0 + krow) * NPIX + n0 + nseg);
}

// Store the 8 bf16 transposed into LDS: Bs[n][k].
__device__ __forceinline__ void commit_B_dense(u16* Bs, uint4 bR, int tid) {
    int krow = tid >> 3;
    int nseg = (tid & 7) * 8;
    union { uint4 q; u16 s[8]; } t; t.q = bR;
#pragma unroll
    for (int j = 0; j < 8; ++j)
        Bs[(nseg + j) * LDP + krow] = t.s[j];
}

// im2col B tile for conv2: gather 8 bf16/thread with zero-pad bounds checks.
__device__ __forceinline__ void issue_B_im2col(const u16* __restrict__ in,
                                               int k0, int n0, int tid, u16* bv) {
#pragma unroll
    for (int j = 0; j < 8; ++j) {
        int e = tid * 8 + j;
        int k = e >> 6, nc = e & 63;
        int kk = k0 + k;
        int ci = kk / 9, r9 = kk % 9;
        int kh = r9 / 3 - 1, kw = r9 % 3 - 1;
        int p = n0 + nc;
        int n = p / 196, hw = p % 196;
        int h = hw / 14 + kh, w = hw % 14 + kw;
        u16 v = 0;
        if ((unsigned)h < 14u && (unsigned)w < 14u)
            v = in[(size_t)ci * NPIX + n * 196 + h * 14 + w];
        bv[j] = v;
    }
}

__device__ __forceinline__ void commit_B_im2col(u16* Bs, const u16* bv, int tid) {
#pragma unroll
    for (int j = 0; j < 8; ++j) {
        int e = tid * 8 + j;
        Bs[(e & 63) * LDP + (e >> 6)] = bv[j];
    }
}

// A fragment, ISA 16-bit A 16x32 layout.
__device__ __forceinline__ v16bf load_frag_a(const u16* As, int mrow, int lane) {
    int half = lane >> 4;
    int base = (mrow + (lane & 15)) * LDP;
    union { uint4 q[2]; v16bf v; } u;
    u.q[0] = *reinterpret_cast<const uint4*>(As + base + half * 8);
    u.q[1] = *reinterpret_cast<const uint4*>(As + base + 16 + half * 8);
    return u.v;
}

// B fragment, ISA 16-bit B 32x16 layout (Bs is N-major in LDS).
__device__ __forceinline__ v16bf load_frag_b(const u16* Bs, int ncol, int lane) {
    int half = lane >> 4;
    int base = (ncol + (lane & 15)) * LDP;
    union { uint4 q[2]; v16bf v; } u;
    u.q[0] = *reinterpret_cast<const uint4*>(Bs + base + half * 8);
    u.q[1] = *reinterpret_cast<const uint4*>(Bs + base + 16 + half * 8);
    return u.v;
}

#define WMMA_BF16(a, b, c) \
    __builtin_amdgcn_wmma_f32_16x16x32_bf16(false, (a), false, (b), (short)0, (c), false, false)

#define COMPUTE_STEP(AsBuf, BsBuf)                                              \
    do {                                                                        \
        v16bf af[2], bf[2];                                                     \
        af[0] = load_frag_a((AsBuf), wm,      lane);                            \
        af[1] = load_frag_a((AsBuf), wm + 16, lane);                            \
        bf[0] = load_frag_b((BsBuf), wn,      lane);                            \
        bf[1] = load_frag_b((BsBuf), wn + 16, lane);                            \
        acc[0][0] = WMMA_BF16(af[0], bf[0], acc[0][0]);                         \
        acc[0][1] = WMMA_BF16(af[0], bf[1], acc[0][1]);                         \
        acc[1][0] = WMMA_BF16(af[1], bf[0], acc[1][0]);                         \
        acc[1][1] = WMMA_BF16(af[1], bf[1], acc[1][1]);                         \
    } while (0)

// ------------------------- conv1: 1x1 s2 (M=256, K=512) --------------------

__global__ void __launch_bounds__(256)
gemm_conv1(const u16* __restrict__ A, const u16* __restrict__ B,
           const float* __restrict__ bias, u16* __restrict__ out) {
    __shared__ __align__(16) u16 As[2][128 * LDP];
    __shared__ __align__(16) u16 Bs[2][64 * LDP];
    const int tid = threadIdx.x, lane = tid & 31, wv = tid >> 5;
    const int wm = (wv >> 1) * 32, wn = (wv & 1) * 32;
    const int m0 = blockIdx.y * 128, n0 = blockIdx.x * 64;
    const int NSTEP = 512 / 32;

    uint4 aR[2]; uint4 bR;
    issue_A(As[0], A, 512, m0, 0, tid, aR);
    bR = issue_B_dense(B, 0, n0, tid);
    commit_A(As[0], tid, aR);
    commit_B_dense(Bs[0], bR, tid);
    async_wait();
    __syncthreads();

    v8f acc[2][2] = {};
    for (int s = 0; s < NSTEP; ++s) {
        int cur = s & 1, nxt = cur ^ 1;
        bool more = (s + 1 < NSTEP);
        if (more) {
            issue_A(As[nxt], A, 512, m0, (s + 1) * 32, tid, aR);
            bR = issue_B_dense(B, (s + 1) * 32, n0, tid);
        }
        COMPUTE_STEP(As[cur], Bs[cur]);
        if (more) {
            commit_A(As[nxt], tid, aR);
            commit_B_dense(Bs[nxt], bR, tid);
            async_wait();
            __syncthreads();
        }
    }

    const int half = lane >> 4, col = lane & 15;
#pragma unroll
    for (int mt = 0; mt < 2; ++mt)
#pragma unroll
        for (int nt = 0; nt < 2; ++nt)
#pragma unroll
            for (int r = 0; r < 8; ++r) {
                int row = m0 + wm + mt * 16 + half * 8 + r;
                int c   = n0 + wn + nt * 16 + col;
                float v = fmaxf(acc[mt][nt][r] + bias[row], 0.0f);
                out[(size_t)row * NPIX + c] = f2bf(v);
            }
}

// ------------------ conv2: 3x3 pad1, implicit im2col (K=2304) --------------

__global__ void __launch_bounds__(256)
gemm_conv2(const u16* __restrict__ A, const u16* __restrict__ in,
           const float* __restrict__ bias, u16* __restrict__ out) {
    __shared__ __align__(16) u16 As[2][128 * LDP];
    __shared__ __align__(16) u16 Bs[2][64 * LDP];
    const int tid = threadIdx.x, lane = tid & 31, wv = tid >> 5;
    const int wm = (wv >> 1) * 32, wn = (wv & 1) * 32;
    const int m0 = blockIdx.y * 128, n0 = blockIdx.x * 64;
    const int NSTEP = 2304 / 32;

    uint4 aR[2]; u16 bv[8];
    issue_A(As[0], A, 2304, m0, 0, tid, aR);
    issue_B_im2col(in, 0, n0, tid, bv);
    commit_A(As[0], tid, aR);
    commit_B_im2col(Bs[0], bv, tid);
    async_wait();
    __syncthreads();

    v8f acc[2][2] = {};
    for (int s = 0; s < NSTEP; ++s) {
        int cur = s & 1, nxt = cur ^ 1;
        bool more = (s + 1 < NSTEP);
        if (more) {
            issue_A(As[nxt], A, 2304, m0, (s + 1) * 32, tid, aR);
            issue_B_im2col(in, (s + 1) * 32, n0, tid, bv);
        }
        COMPUTE_STEP(As[cur], Bs[cur]);
        if (more) {
            commit_A(As[nxt], tid, aR);
            commit_B_im2col(Bs[nxt], bv, tid);
            async_wait();
            __syncthreads();
        }
    }

    const int half = lane >> 4, col = lane & 15;
#pragma unroll
    for (int mt = 0; mt < 2; ++mt)
#pragma unroll
        for (int nt = 0; nt < 2; ++nt)
#pragma unroll
            for (int r = 0; r < 8; ++r) {
                int row = m0 + wm + mt * 16 + half * 8 + r;
                int c   = n0 + wn + nt * 16 + col;
                float v = fmaxf(acc[mt][nt][r] + bias[row], 0.0f);
                out[(size_t)row * NPIX + c] = f2bf(v);
            }
}

// -------- conv3 (K=256) + shortcut conv4 (K=512) fused, f32 NCHW out -------

__global__ void __launch_bounds__(256)
gemm_conv3(const u16* __restrict__ A3, const u16* __restrict__ B3,
           const u16* __restrict__ A4, const u16* __restrict__ B4,
           const float* __restrict__ b3, const float* __restrict__ b4,
           float* __restrict__ out) {
    __shared__ __align__(16) u16 As[2][128 * LDP];
    __shared__ __align__(16) u16 Bs[2][64 * LDP];
    const int tid = threadIdx.x, lane = tid & 31, wv = tid >> 5;
    const int wm = (wv >> 1) * 32, wn = (wv & 1) * 32;
    const int m0 = blockIdx.y * 128, n0 = blockIdx.x * 64;
    const int NSTEP = 8 + 16;   // 256/32 steps of w3 x out2, then 512/32 of w4 x xs

    auto params = [&](int s, const u16*& Ap, const u16*& Bp, int& K, int& k0) {
        if (s < 8) { Ap = A3; Bp = B3; K = 256; k0 = s * 32; }
        else       { Ap = A4; Bp = B4; K = 512; k0 = (s - 8) * 32; }
    };

    const u16 *Ap, *Bp; int K, k0;
    uint4 aR[2]; uint4 bR;
    params(0, Ap, Bp, K, k0);
    issue_A(As[0], Ap, K, m0, k0, tid, aR);
    bR = issue_B_dense(Bp, k0, n0, tid);
    commit_A(As[0], tid, aR);
    commit_B_dense(Bs[0], bR, tid);
    async_wait();
    __syncthreads();

    v8f acc[2][2] = {};
    for (int s = 0; s < NSTEP; ++s) {
        int cur = s & 1, nxt = cur ^ 1;
        bool more = (s + 1 < NSTEP);
        if (more) {
            params(s + 1, Ap, Bp, K, k0);
            issue_A(As[nxt], Ap, K, m0, k0, tid, aR);
            bR = issue_B_dense(Bp, k0, n0, tid);
        }
        COMPUTE_STEP(As[cur], Bs[cur]);
        if (more) {
            commit_A(As[nxt], tid, aR);
            commit_B_dense(Bs[nxt], bR, tid);
            async_wait();
            __syncthreads();
        }
    }

    const int half = lane >> 4, col = lane & 15;
#pragma unroll
    for (int mt = 0; mt < 2; ++mt)
#pragma unroll
        for (int nt = 0; nt < 2; ++nt)
#pragma unroll
            for (int r = 0; r < 8; ++r) {
                int row = m0 + wm + mt * 16 + half * 8 + r;   // out channel
                int c   = n0 + wn + nt * 16 + col;            // pixel
                float v = fmaxf(acc[mt][nt][r] + b3[row] + b4[row], 0.0f);
                int n = c / 196, hw = c % 196;
                out[(size_t)n * (COUT * 196) + (size_t)row * 196 + hw] = v;
            }
}

// ------------------------------- launch ------------------------------------

extern "C" void kernel_launch(void* const* d_in, const int* in_sizes, int n_in,
                              void* d_out, int out_size, void* d_ws, size_t ws_size,
                              hipStream_t stream) {
    const float* x  = (const float*)d_in[0];
    const float* w1 = (const float*)d_in[1];
    const float* w2 = (const float*)d_in[2];
    const float* w3 = (const float*)d_in[3];
    const float* w4 = (const float*)d_in[4];
    const float* s1 = (const float*)d_in[5];
    const float* b1 = (const float*)d_in[6];
    const float* s2 = (const float*)d_in[7];
    const float* b2 = (const float*)d_in[8];
    const float* s3 = (const float*)d_in[9];
    const float* b3 = (const float*)d_in[10];
    const float* s4 = (const float*)d_in[11];
    const float* b4 = (const float*)d_in[12];

    char* ws = (char*)d_ws;
    u16* xs   = (u16*)(ws + 0);            // 512  x 12544 bf16
    u16* out1 = (u16*)(ws + 12845056);     // 256  x 12544 bf16
    u16* out2 = (u16*)(ws + 19267584);     // 256  x 12544 bf16
    u16* w1b  = (u16*)(ws + 25690112);     // 256  x 512
    u16* w2b  = (u16*)(ws + 25952256);     // 256  x 2304
    u16* w3b  = (u16*)(ws + 27131904);     // 1024 x 256
    u16* w4b  = (u16*)(ws + 27656192);     // 1024 x 512

    scale_w_kernel<<<(256 * 512 + 255) / 256, 256, 0, stream>>>(w1, s1, w1b, 256, 512);
    scale_w_kernel<<<(256 * 2304 + 255) / 256, 256, 0, stream>>>(w2, s2, w2b, 256, 2304);
    scale_w_kernel<<<(1024 * 256 + 255) / 256, 256, 0, stream>>>(w3, s3, w3b, 1024, 256);
    scale_w_kernel<<<(1024 * 512 + 255) / 256, 256, 0, stream>>>(w4, s4, w4b, 1024, 512);
    gather_x_kernel<<<(CIN * NPIX + 255) / 256, 256, 0, stream>>>(x, xs);

    gemm_conv1<<<dim3(196, 2), 256, 0, stream>>>(w1b, xs, b1, out1);
    gemm_conv2<<<dim3(196, 2), 256, 0, stream>>>(w2b, out1, b2, out2);
    gemm_conv3<<<dim3(196, 8), 256, 0, stream>>>(w3b, out2, w4b, xs, b3, b4, (float*)d_out);
}
